// QLSTM_65481071395271
// MI455X (gfx1250) — matrix operand order repaired and checked by reading, not compiled, not run on hardware
//
#include <hip/hip_runtime.h>
#include <hip/hip_bf16.h>

// ---------------------------------------------------------------------------
// LSTM: T=256, B=64, D=512, H=1024.  Per step: gates = [x|h] @ W_all^T + b,
// then elementwise cell update.  MI455X (gfx1250) strategy:
//  * bf16 weights (12.6 MB -> L2 resident), v_wmma_f32_16x16x32_bf16
//  * persistent kernel, 64 WGs x 512 thr (16 waves), one 16x16 tile per wave
//  * h_prev staged into LDS via GLOBAL_LOAD_ASYNC_TO_LDS_B128, overlapped
//    with the x-part of the GEMM (ASYNCcnt / s_wait_asynccnt)
//  * grid barrier per timestep (monotonic atomic counter), ping-pong h bufs
//  * cell state c lives in registers for the whole 256-step loop
// ---------------------------------------------------------------------------

#define TSTEPS 256
#define BATCH  64
#define DIN    512
#define HID    1024
#define KTOT   (DIN + HID)        // 1536
#define NBLK   (HID / 16)         // 64 workgroups
#define NTHR   512                // 16 waves of 32
#define PSTRIDE 20                // preact LDS row stride (floats)
#define HROW   (HID + 8)          // staged-h LDS row stride (bf16): 2064 B
                                  // -> 4-dword bank shift/row, conflict-light

typedef __attribute__((ext_vector_type(16))) __bf16 v16bf;
typedef __attribute__((ext_vector_type(8)))  __bf16 v8bf;
typedef __attribute__((ext_vector_type(8)))  float  v8f;
typedef __attribute__((ext_vector_type(4)))  int    v4i;

#if defined(__gfx1250__) && __has_builtin(__builtin_amdgcn_global_load_async_to_lds_b128)
#define HAVE_ASYNC_LDS 1
#else
#define HAVE_ASYNC_LDS 0
#endif

// ---- helpers --------------------------------------------------------------

__device__ __forceinline__ float sigm(float x) {
    return 1.0f / (1.0f + __expf(-x));
}
__device__ __forceinline__ float tanh_fast(float x) {
    float e = __expf(2.0f * x);          // saturates correctly at +/-1
    return 1.0f - 2.0f / (e + 1.0f);
}

// A fragment (16x32 bf16): lane holds row m=lane&15, K groups {klo..klo+7} and
// {klo+16..klo+23}, klo = (lane<16)?0:8.  Caller bakes klo into rowPtr.
__device__ __forceinline__ v16bf load_a16(const __bf16* p) {
    v8bf lo = *(const v8bf*)(p);
    v8bf hi = *(const v8bf*)(p + 16);
    return __builtin_shufflevector(lo, hi, 0,1,2,3,4,5,6,7,8,9,10,11,12,13,14,15);
}

__device__ __forceinline__ void wait_async_done() {
#if HAVE_ASYNC_LDS
#if __has_builtin(__builtin_amdgcn_s_wait_asynccnt)
    __builtin_amdgcn_s_wait_asynccnt(0);
#else
    asm volatile("s_wait_asynccnt 0x0" ::: "memory");
#endif
#endif
}

// ---- prep kernels ---------------------------------------------------------

__global__ void f32_to_bf16_kernel(const float* __restrict__ src,
                                   __bf16* __restrict__ dst, int n) {
    int i = blockIdx.x * blockDim.x + threadIdx.x;
    if (i < n) dst[i] = (__bf16)src[i];
}

__global__ void init_state_kernel(__bf16* __restrict__ hbuf0,
                                  unsigned* __restrict__ counter) {
    int i = blockIdx.x * blockDim.x + threadIdx.x;
    if (i == 0) *counter = 0u;
    if (i < BATCH * HID) hbuf0[i] = (__bf16)0.0f;
}

// ---- persistent LSTM kernel ----------------------------------------------

__global__ __launch_bounds__(NTHR, 1)
void lstm_persistent(const __bf16* __restrict__ Wbf,   // [4][HID][KTOT] bf16
                     const __bf16* __restrict__ Xbf,   // [T][B][DIN]   bf16
                     const float*  __restrict__ bfp,
                     const float*  __restrict__ bip,
                     const float*  __restrict__ bgp,
                     const float*  __restrict__ bop,
                     __bf16* __restrict__ hbuf0,       // [B][HID] ping
                     __bf16* __restrict__ hbuf1,       // [B][HID] pong
                     float*  __restrict__ out,         // [T][B][HID] ++ hT ++ cT
                     unsigned* __restrict__ counter)
{
    __shared__ __align__(16) __bf16 ldsH[BATCH * HROW];     // 129 KB staged h
    __shared__ float preact[4 * BATCH * PSTRIDE];           // 20 KB

    const int tid   = threadIdx.x;
    const int lane  = tid & 31;
    const int wave  = tid >> 5;          // 0..15
    const int gate  = wave & 3;          // 0=f 1=i 2=g 3=o
    const int mTile = wave >> 2;         // 0..3  (batch 16-row tile)
    const int nBase = blockIdx.x * 16;   // hidden column base of this WG

    // WMMA fragment lane addressing (per ISA 16-bit A/B layouts)
    const int mA    = mTile * 16 + (lane & 15);
    const int kloA  = (lane & 16) ? 8 : 0;
    const int nCol  = nBase + (lane & 15);
    const int kloB  = (lane & 16) ? 8 : 0;
    const __bf16* wRow = Wbf + (size_t)(gate * HID + nCol) * KTOT + kloB;

    const float* bp = (gate == 0) ? bfp : (gate == 1) ? bip
                    : (gate == 2) ? bgp : bop;
    const float bias = bp[nCol];

    // staged-h A fragment base (generic ptr into LDS -> ds_load_b128)
    const __bf16* aRowH = &ldsH[(size_t)mA * HROW + kloA];

    // Elementwise stage: thread owns 2 (m, nLocal) pairs; c stays in VGPRs.
    const int m0 = tid >> 4,          nl0 = tid & 15;          // m 0..31
    const int m1 = (tid + NTHR) >> 4, nl1 = tid & 15;          // m 32..63
    float c0 = 0.0f, c1 = 0.0f;

    for (int t = 0; t < TSTEPS; ++t) {
        const __bf16* hprev = (t & 1) ? hbuf1 : hbuf0;
        __bf16*       hnext = (t & 1) ? hbuf0 : hbuf1;

        // ---- stage h_prev [64][1024] bf16 -> LDS, async, 16B chunks ------
        // 8192 chunks over 512 threads = 16 chunks each.
        #pragma unroll
        for (int i = 0; i < 16; ++i) {
            const int c = tid + NTHR * i;        // chunk id
            const int m = c >> 7;                // row (128 chunks/row)
            const int e = (c & 127) << 3;        // element in row (8 bf16/chunk)
            const __bf16* g = hprev + (size_t)m * HID + e;
            __bf16*       l = &ldsH[(size_t)m * HROW + e];
#if HAVE_ASYNC_LDS
            __builtin_amdgcn_global_load_async_to_lds_b128(
                (v4i*)g, (v4i*)l, 0, 0);
#else
            *(v8bf*)l = *(const v8bf*)g;
#endif
        }

        // ---- x-part GEMM (K = 0..511), overlaps with the staging ---------
        v8f acc = {};
        const __bf16* aRowX = Xbf + ((size_t)t * BATCH + mA) * DIN + kloA;
        #pragma unroll 4
        for (int k = 0; k < DIN; k += 32) {
            v16bf a = load_a16(aRowX + k);
            v16bf b = load_a16(wRow + k);
            acc = __builtin_amdgcn_wmma_f32_16x16x32_bf16(
                    false, a, false, b, (short)0, acc, false, false);
        }

        // ---- staged h ready for the whole block --------------------------
        wait_async_done();
        __syncthreads();

        // ---- h-part GEMM (K = 512..1535), A from LDS ---------------------
        #pragma unroll 4
        for (int k = 0; k < HID; k += 32) {
            v16bf a = load_a16(aRowH + k);
            v16bf b = load_a16(wRow + DIN + k);
            acc = __builtin_amdgcn_wmma_f32_16x16x32_bf16(
                    false, a, false, b, (short)0, acc, false, false);
        }

        // ---- scatter pre-activations (+bias) to LDS ----------------------
        // C/D layout: VGPR r, lanes 0-15 -> M=r, lanes 16-31 -> M=r+8; N=lane&15
        {
            const int mOff = mTile * 16 + ((lane & 16) >> 1); // +8 upper half
            #pragma unroll
            for (int r = 0; r < 8; ++r) {
                preact[(gate * BATCH + (mOff + r)) * PSTRIDE + (lane & 15)]
                    = acc[r] + bias;
            }
        }
        __syncthreads();

        // ---- fused gate math; c in registers -----------------------------
        {
            float fp = preact[(0 * BATCH + m0) * PSTRIDE + nl0];
            float ip = preact[(1 * BATCH + m0) * PSTRIDE + nl0];
            float gp = preact[(2 * BATCH + m0) * PSTRIDE + nl0];
            float op = preact[(3 * BATCH + m0) * PSTRIDE + nl0];
            c0 = sigm(fp) * c0 + sigm(ip) * tanh_fast(gp);
            float h = sigm(op) * tanh_fast(c0);
            hnext[m0 * HID + nBase + nl0] = (__bf16)h;
            out[((size_t)t * BATCH + m0) * HID + nBase + nl0] = h;
            if (t == TSTEPS - 1) {
                out[(size_t)TSTEPS * BATCH * HID + m0 * HID + nBase + nl0] = h;
                out[(size_t)TSTEPS * BATCH * HID + BATCH * HID
                    + m0 * HID + nBase + nl0] = c0;
            }
        }
        {
            float fp = preact[(0 * BATCH + m1) * PSTRIDE + nl1];
            float ip = preact[(1 * BATCH + m1) * PSTRIDE + nl1];
            float gp = preact[(2 * BATCH + m1) * PSTRIDE + nl1];
            float op = preact[(3 * BATCH + m1) * PSTRIDE + nl1];
            c1 = sigm(fp) * c1 + sigm(ip) * tanh_fast(gp);
            float h = sigm(op) * tanh_fast(c1);
            hnext[m1 * HID + nBase + nl1] = (__bf16)h;
            out[((size_t)t * BATCH + m1) * HID + nBase + nl1] = h;
            if (t == TSTEPS - 1) {
                out[(size_t)TSTEPS * BATCH * HID + m1 * HID + nBase + nl1] = h;
                out[(size_t)TSTEPS * BATCH * HID + BATCH * HID
                    + m1 * HID + nBase + nl1] = c1;
            }
        }

        // ---- grid-wide barrier (monotonic counter, agent scope) ----------
        __threadfence();                 // release h stores to device scope
        __syncthreads();
        if (tid == 0) {
            __hip_atomic_fetch_add(counter, 1u, __ATOMIC_ACQ_REL,
                                   __HIP_MEMORY_SCOPE_AGENT);
            const unsigned target = (unsigned)(t + 1) * (unsigned)gridDim.x;
            while (__hip_atomic_load(counter, __ATOMIC_ACQUIRE,
                                     __HIP_MEMORY_SCOPE_AGENT) < target) {
                __builtin_amdgcn_s_sleep(2);
            }
        }
        __syncthreads();
    }
}

// ---------------------------------------------------------------------------

extern "C" void kernel_launch(void* const* d_in, const int* in_sizes, int n_in,
                              void* d_out, int out_size, void* d_ws, size_t ws_size,
                              hipStream_t stream) {
    const float* inputs = (const float*)d_in[0];   // [T,B,D]
    const float* Wf = (const float*)d_in[1];
    const float* bf = (const float*)d_in[2];
    const float* Wi = (const float*)d_in[3];
    const float* bi = (const float*)d_in[4];
    const float* Wg = (const float*)d_in[5];
    const float* bg = (const float*)d_in[6];
    const float* Wo = (const float*)d_in[7];
    const float* bo = (const float*)d_in[8];
    float* out = (float*)d_out;

    // workspace layout
    __bf16* Wbf   = (__bf16*)d_ws;                       // [4][HID][KTOT]
    __bf16* Xbf   = Wbf + (size_t)4 * HID * KTOT;        // [T][B][DIN]
    __bf16* hbuf0 = Xbf + (size_t)TSTEPS * BATCH * DIN;  // [B][HID]
    __bf16* hbuf1 = hbuf0 + (size_t)BATCH * HID;
    unsigned* counter = (unsigned*)(hbuf1 + (size_t)BATCH * HID);

    const int nW = HID * KTOT;                           // per-gate weight count
    const float* Ws[4] = { Wf, Wi, Wg, Wo };
    for (int g = 0; g < 4; ++g) {
        f32_to_bf16_kernel<<<(nW + 255) / 256, 256, 0, stream>>>(
            Ws[g], Wbf + (size_t)g * nW, nW);
    }
    const int nX = TSTEPS * BATCH * DIN;
    f32_to_bf16_kernel<<<(nX + 255) / 256, 256, 0, stream>>>(inputs, Xbf, nX);

    init_state_kernel<<<(BATCH * HID + 255) / 256, 256, 0, stream>>>(hbuf0, counter);

    lstm_persistent<<<NBLK, NTHR, 0, stream>>>(
        Wbf, Xbf, bf, bi, bg, bo, hbuf0, hbuf1, out, counter);
}
